// ResidualAttentionBlock_28862180229630
// MI455X (gfx1250) — compile-verified, matrix operand order checked by reading, not run on hardware
//
#include <hip/hip_runtime.h>
#include <hip/hip_bf16.h>

// MI455X / gfx1250 implementation.
// Workspace (17 MB): qh = f16[16][8][512][64], vh same, xh = f16[16][512][64] (padded).
//
// Kernel 1 (qv_proj): WMMA f16 GEMM for q/v projections (M=B*512, N=1024, K=64),
//                     also emits the padded f16 copy of x (xh) for kernel 2.
// Kernel 2 (attn): per-(batch, 16-query tile) block, 8 waves = 8 heads.
//   - x key-chunks staged with global_load_async_to_lds_b128 (ASYNCcnt), double-buffered
//   - packed-f16 VALU for the L1 distance: |fma(x, wk, -q)| -> v_pk_fma/and/v_pk_add
//   - softmax max is provably 0 (logits <= 0, sink logit = 0) -> denom = 1 + sum(exp)
//   - e (x) v accumulated with v_wmma_f32_16x16x32_f16
//   - head merge via LDS ds_add_f32, then ReLU + fanout WMMA + residual.

typedef __attribute__((ext_vector_type(16))) _Float16 v16h;
typedef __attribute__((ext_vector_type(8)))  float    v8f;
typedef _Float16 h2 __attribute__((ext_vector_type(2)));

#define B_   16
#define T_   511
#define TP_  512
#define D_   64
#define H_   8

static __device__ inline h2 habs2(h2 v) {
  unsigned u = __builtin_bit_cast(unsigned, v);
  u &= 0x7FFF7FFFu;
  return __builtin_bit_cast(h2, u);
}

// ---------------------------------------------------------------------------
// Kernel 1: q/v projections + padded f16 copy of x.
// grid.x = (B_*TP_)/16 = 512 blocks, 256 threads (8 waves).
// ---------------------------------------------------------------------------
__global__ __launch_bounds__(256) void qv_proj_kernel(
    const float* __restrict__ x, const float* __restrict__ wq,
    const float* __restrict__ wv, _Float16* __restrict__ qh,
    _Float16* __restrict__ vh, _Float16* __restrict__ xh) {
  __shared__ _Float16 xs[16][64];
  const int lid  = threadIdx.x;
  const int lane = lid & 31;
  const int mtile = blockIdx.x;

  // Each thread handles 4 consecutive elements of the 16x64 tile:
  // x (f32, guarded) -> f16 into LDS and into the padded global copy xh.
  {
    int i4 = lid * 4;
    int m = i4 >> 6, w = i4 & 63;
    int g = mtile * 16 + m;
    int b = g >> 9, t = g & 511;
    float4 v = make_float4(0.f, 0.f, 0.f, 0.f);
    if (t < T_) v = *(const float4*)&x[((size_t)b * T_ + t) * D_ + w];
    h2 lo = { (_Float16)v.x, (_Float16)v.y };
    h2 hi = { (_Float16)v.z, (_Float16)v.w };
    ((h2*)&xs[m][w])[0] = lo;
    ((h2*)&xs[m][w])[1] = hi;
    h2* xrow = (h2*)(xh + (size_t)g * D_ + w);
    xrow[0] = lo;
    xrow[1] = hi;
  }
  __syncthreads();

  const int mrow = lane & 15;          // A-fragment row (M)
  const int hb   = (lane >> 4) << 3;   // K-half base per ISA 16-bit A layout

  // A fragments for K = 0..31 and K = 32..63 (shared by all 8 N-tiles).
  v16h a0, a1;
#pragma unroll
  for (int e = 0; e < 16; ++e) {
    int kk = ((e >> 3) << 4) + hb + (e & 7);   // element e <-> K index
    a0[e] = xs[mrow][kk];
    a1[e] = xs[mrow][32 + kk];
  }

  const int wid = lid >> 5;
  const int n   = lane & 15;           // B-fragment column (N)
  for (int i = 0; i < 8; ++i) {
    int ot = wid * 8 + i;              // 0..31 -> q, 32..63 -> v
    const float* wmat = (ot < 32) ? wq : wv;
    _Float16*    dst  = (ot < 32) ? qh : vh;
    int obase = (ot & 31) * 16;
    int o = obase + n;
    const float* wrow = wmat + (size_t)o * 65;   // row-major [o][k], col 64 = bias

    v16h b0, b1;
#pragma unroll
    for (int e = 0; e < 16; ++e) {
      int kk = ((e >> 3) << 4) + hb + (e & 7);
      b0[e] = (_Float16)wrow[kk];        // B[k][n] = w[o=n][k]
      b1[e] = (_Float16)wrow[32 + kk];
    }
    v8f c = {};
    c = __builtin_amdgcn_wmma_f32_16x16x32_f16(false, a0, false, b0, (short)0, c, false, false);
    c = __builtin_amdgcn_wmma_f32_16x16x32_f16(false, a1, false, b1, (short)0, c, false, false);

    float bias = wrow[64];
    int hh = o >> 6, wcol = o & 63;
#pragma unroll
    for (int r = 0; r < 8; ++r) {
      int m = r + ((lane >> 4) << 3);    // C layout: lanes 16-31 hold M = r+8
      int g = mtile * 16 + m;
      int bb = g >> 9, t = g & 511;
      dst[(((size_t)bb * H_ + hh) * TP_ + t) * D_ + wcol] = (_Float16)(c[r] + bias);
    }
  }
}

// ---------------------------------------------------------------------------
// Kernel 2: attention + head-merge + ReLU + fanout + residual.
// grid = (TP_/16 = 32 d-tiles, B_), 256 threads = 8 waves, wave h = head h.
// ---------------------------------------------------------------------------
__global__ __launch_bounds__(256) void attn_kernel(
    const float* __restrict__ x, const float* __restrict__ wk,
    const float* __restrict__ fw, const _Float16* __restrict__ qh,
    const _Float16* __restrict__ vh, const _Float16* __restrict__ xh,
    float* __restrict__ out) {
  __shared__ _Float16 xs[2][32][64];  // double-buffered key-chunk (async-to-LDS)
  __shared__ float oacc[16][64];      // head-merged numerator tile
  __shared__ float dens[H_][16];      // per-head per-row softmax denominators

  const int lid  = threadIdx.x;
  const int lane = lid & 31;
  const int h    = lid >> 5;          // head = wave id
  const int dtile = blockIdx.x;
  const int b     = blockIdx.y;
  const int m  = lane & 15;           // fragment row/column index
  const int hb = (lane >> 4) << 3;    // K-half base

  for (int i = lid; i < 16 * 64; i += 256) ((float*)oacc)[i] = 0.0f;

  // Per-lane -q row (query d = dtile*16 + m, head h) and wk row, as 32 packed half2.
  h2 qnr[32], wkr[32];
  {
    const _Float16* qrow = qh + (((size_t)b * H_ + h) * TP_ + (dtile * 16 + m)) * D_;
    const float* wkrow = wk + (size_t)h * D_;
#pragma unroll
    for (int j = 0; j < 32; ++j) {
      qnr[j] = -((const h2*)qrow)[j];
      h2 t;
      t.x = (_Float16)wkrow[2 * j];
      t.y = (_Float16)wkrow[2 * j + 1];
      wkr[j] = t;
    }
  }

  // Async stage one 32-key chunk (4 KB) into LDS: 256 threads x one b128 each.
  const _Float16* xbat = xh + (size_t)b * TP_ * D_;
  auto issue_chunk = [&](int sc) {
    unsigned ldsoff = (unsigned)(size_t)(&xs[sc & 1][0][0]) + (unsigned)lid * 16u;
    unsigned long long gaddr =
        (unsigned long long)(size_t)(xbat + (size_t)sc * 32 * D_ + (size_t)lid * 8);
    asm volatile("global_load_async_to_lds_b128 %0, %1, off"
                 :: "v"(ldsoff), "v"(gaddr) : "memory");
  };

  v8f n0 = {}, n1 = {}, n2 = {}, n3 = {};
  float denp = 0.0f;

  issue_chunk(0);
  for (int sc = 0; sc < 16; ++sc) {           // 16 chunks of 32 keys
    const int sbase = sc * 32;
    asm volatile("s_wait_asynccnt 0x0" ::: "memory");
    __syncthreads();                           // chunk sc visible to all waves
    if (sc + 1 < 16) issue_chunk(sc + 1);      // overlap next copy with compute
    const _Float16 (*xcur)[64] = xs[sc & 1];

    // exp(-(1/8) * sum_w |x*wk - q|) for the 16 (m,s) pairs this lane owns
    // (exactly the A-fragment slots of the upcoming WMMA).
    v16h afrag;
#pragma unroll
    for (int e = 0; e < 16; ++e) {
      int sl = ((e >> 3) << 4) + hb + (e & 7);   // element e <-> K (= key) index
      const h2* xrow = (const h2*)xcur[sl];
      h2 acc2 = { (_Float16)0, (_Float16)0 };
#pragma unroll
      for (int j = 0; j < 32; ++j) {
        h2 d = xrow[j] * wkr[j] + qnr[j];         // v_pk_fma_f16
        acc2 += habs2(d);                         // and-mask + v_pk_add_f16
      }
      float sum = (float)acc2.x + (float)acc2.y;
      float ev = __expf(-0.125f * sum);           // logits <= 0, max is the sink's 0
      if (sbase + sl >= T_) ev = 0.0f;            // mask pad key
      denp += ev;
      afrag[e] = (_Float16)ev;
    }

    // B fragments from v (4 column-tiles of 16), then accumulate e (x) v.
    const _Float16* vbase = vh + (((size_t)b * H_ + h) * TP_ + sbase) * D_;
    v16h b0, b1, b2, b3;
#pragma unroll
    for (int e = 0; e < 16; ++e) {
      int sl = ((e >> 3) << 4) + hb + (e & 7);
      const _Float16* vr = vbase + sl * D_ + m;
      b0[e] = vr[0];
      b1[e] = vr[16];
      b2[e] = vr[32];
      b3[e] = vr[48];
    }
    n0 = __builtin_amdgcn_wmma_f32_16x16x32_f16(false, afrag, false, b0, (short)0, n0, false, false);
    n1 = __builtin_amdgcn_wmma_f32_16x16x32_f16(false, afrag, false, b1, (short)0, n1, false, false);
    n2 = __builtin_amdgcn_wmma_f32_16x16x32_f16(false, afrag, false, b2, (short)0, n2, false, false);
    n3 = __builtin_amdgcn_wmma_f32_16x16x32_f16(false, afrag, false, b3, (short)0, n3, false, false);
  }

  // Per-row denominator: lanes L and L^16 share row m and cover disjoint keys.
  denp += __shfl_xor(denp, 16, 32);
  if (lane < 16) dens[h][m] = 1.0f + denp;       // +1 = sink key exp(0)
  __syncthreads();

  // Head merge: out[m][w] += N_h[m][w] / den_h[m]   (ds_add_f32 in LDS)
#pragma unroll
  for (int r = 0; r < 8; ++r) {
    int mr = r + ((lane >> 4) << 3);
    float inv = 1.0f / dens[h][mr];
    atomicAdd(&oacc[mr][ 0 + m], n0[r] * inv);
    atomicAdd(&oacc[mr][16 + m], n1[r] * inv);
    atomicAdd(&oacc[mr][32 + m], n2[r] * inv);
    atomicAdd(&oacc[mr][48 + m], n3[r] * inv);
  }
  __syncthreads();

  // Fanout: z = relu(oacc) @ fw[:, :64]^T + fw[:, 64]; out = x + z.  Waves 0..3.
  if (h < 4) {
    const int jbase = h * 16;
    const float* frow = fw + (size_t)(jbase + m) * 65;
    v16h ya0, ya1, fb0, fb1;
#pragma unroll
    for (int e = 0; e < 16; ++e) {
      int kk = ((e >> 3) << 4) + hb + (e & 7);
      float y0 = oacc[m][kk];
      float y1 = oacc[m][32 + kk];
      ya0[e] = (_Float16)(y0 > 0.0f ? y0 : 0.0f);
      ya1[e] = (_Float16)(y1 > 0.0f ? y1 : 0.0f);
      fb0[e] = (_Float16)frow[kk];          // B[k][n=j] = fw[j][k]
      fb1[e] = (_Float16)frow[32 + kk];
    }
    v8f c = {};
    c = __builtin_amdgcn_wmma_f32_16x16x32_f16(false, ya0, false, fb0, (short)0, c, false, false);
    c = __builtin_amdgcn_wmma_f32_16x16x32_f16(false, ya1, false, fb1, (short)0, c, false, false);

    float bias = frow[64];
#pragma unroll
    for (int r = 0; r < 8; ++r) {
      int mr = r + ((lane >> 4) << 3);
      int d = dtile * 16 + mr;
      if (d < T_) {
        size_t idx = ((size_t)b * T_ + d) * D_ + jbase + m;
        out[idx] = x[idx] + c[r] + bias;
      }
    }
  }
}

// ---------------------------------------------------------------------------
extern "C" void kernel_launch(void* const* d_in, const int* in_sizes, int n_in,
                              void* d_out, int out_size, void* d_ws, size_t ws_size,
                              hipStream_t stream) {
  (void)in_sizes; (void)n_in; (void)out_size; (void)ws_size;
  const float* x  = (const float*)d_in[0];   // [16][511][64]
  const float* wq = (const float*)d_in[1];   // [512][65]
  const float* wv = (const float*)d_in[2];   // [512][65]
  const float* wk = (const float*)d_in[3];   // [8][64]
  const float* fw = (const float*)d_in[4];   // [64][65]
  float* out = (float*)d_out;                // [16][511][64]

  _Float16* qh = (_Float16*)d_ws;                                  // 8 MB
  _Float16* vh = qh + (size_t)B_ * H_ * TP_ * D_;                  // 8 MB
  _Float16* xh = vh + (size_t)B_ * H_ * TP_ * D_;                  // 1 MB (padded x, f16)

  qv_proj_kernel<<<dim3((B_ * TP_) / 16), 256, 0, stream>>>(x, wq, wv, qh, vh, xh);
  attn_kernel<<<dim3(TP_ / 16, B_), 256, 0, stream>>>(x, wk, fw, qh, vh, xh, out);
}
